// Attention_6347961664085
// MI455X (gfx1250) — matrix operand order repaired and checked by reading, not compiled, not run on hardware
//
#include <hip/hip_runtime.h>
#include <hip/hip_bf16.h>
#include <stdint.h>

// ---------------------------------------------------------------------------
// Types / helpers for CDNA5 WMMA (wave32, 16x16x32 bf16 -> f32)
// ---------------------------------------------------------------------------
typedef __attribute__((ext_vector_type(16))) __bf16 v16bf;
typedef __attribute__((ext_vector_type(8)))  float  v8f;
typedef __attribute__((ext_vector_type(4)))  unsigned int v4u;
typedef __attribute__((ext_vector_type(8)))  unsigned int v8u;

union FragU {
  uint4 q[2];
  unsigned short s[16];
  v16bf v;
};

__device__ __forceinline__ unsigned short f2bf(float f) {
  unsigned int u = __float_as_uint(f);
  u += 0x7FFFu + ((u >> 16) & 1u);          // round-to-nearest-even
  return (unsigned short)(u >> 16);
}

__device__ __forceinline__ v8f v8f_zero() {
  v8f z = {0.f, 0.f, 0.f, 0.f, 0.f, 0.f, 0.f, 0.f};
  return z;
}

// A-operand (16x32 bf16): lanes 0-15 = rows 0-15 (K half 0), lanes 16-31 =
// rows 0-15 (K half 1). Two 16B chunks per lane -> 2x ds_load_b128.
__device__ __forceinline__ v16bf fragA16(const unsigned short* base, int rs,
                                         int kcoff, int lane) {
  FragU f;
  const unsigned short* p = base + (lane & 15) * rs + kcoff + ((lane >> 4) << 3);
  f.q[0] = *(const uint4*)(p);
  f.q[1] = *(const uint4*)(p + 16);
  return f.v;
}

// B-operand (32x16 bf16): lane = column n (lane&15), elements = K in table
// order (k = 8h+s, s<8; 16+8h+s-8, s>=8). Strided u16 gather from a
// row-major K x N tile.
__device__ __forceinline__ v16bf fragB16(const unsigned short* base, int rs,
                                         int kcoff, int ncol, int lane) {
  FragU f;
  const unsigned short* p = base + ncol + (lane & 15);
#pragma unroll
  for (int s = 0; s < 16; ++s) {
    int k = kcoff + ((s < 8) ? 0 : 16) + ((lane >> 4) << 3) + (s & 7);
    f.s[s] = p[k * rs];
  }
  return f.v;
}

// A-operand gathered from a TRANSPOSED store (tile kept as K x M row-major,
// e.g. P^T): strided u16 gather (only used 2x per attention k-block).
__device__ __forceinline__ v16bf fragAT16(const unsigned short* baseT, int rs,
                                          int kcoff, int lane) {
  FragU f;
  const int q = lane & 15, khalf = lane >> 4;
#pragma unroll
  for (int i = 0; i < 8; ++i) {
    int kb = kcoff + ((i < 4) ? 0 : 16) + (khalf << 3) + ((i & 3) << 1);
    f.s[2 * i]     = baseT[kb * rs + q];
    f.s[2 * i + 1] = baseT[(kb + 1) * rs + q];
  }
  return f.v;
}

// B-operand of Q^T gathered from global row-major Q (once per kernel):
// lane&15 = query, elements = K(=d) in table order -> 2x b128 loads.
__device__ __forceinline__ v16bf fragBT16g(const unsigned short* base, int ld,
                                           int dbase, int lane) {
  FragU f;
  const unsigned short* p = base + (size_t)(lane & 15) * ld + dbase + ((lane >> 4) << 3);
  f.q[0] = *(const uint4*)(p);
  f.q[1] = *(const uint4*)(p + 16);
  return f.v;
}

__device__ __forceinline__ v8f wmma_bf16(v16bf a, v16bf b, v8f c) {
  return __builtin_amdgcn_wmma_f32_16x16x32_bf16(false, a, false, b, (short)0, c,
                                                 false, false);
}

// ---------------------------------------------------------------------------
// Tensor Data Mover: async 2D bf16 tile load global -> LDS with row padding.
// Descriptor per CDNA5 ISA ch.8 (group0 128b + group1 256b, 2D form).
//  - padIntCode: DWORDs/row before pad (3=16DW/64B, 4=32DW/128B, 5=64DW/256B)
//  - padAmtCode: 3 = 4 DWORDs = 16 bytes pad
// Issued by a single wave; tracked with TENSORcnt.
// ---------------------------------------------------------------------------
__device__ __forceinline__ void tdm_load_2d(const void* gsrc, unsigned ldsOff,
                                            unsigned tileW, unsigned tileH,
                                            unsigned strideElems,
                                            unsigned padIntCode,
                                            unsigned padAmtCode) {
  unsigned long long ga = (unsigned long long)(size_t)gsrc;
  const unsigned TD0 = 1u << 20, TD1 = 1u << 20;  // huge tensor dims: never OOB
  v4u g0;
  g0[0] = 1u;                                              // count=1, user desc
  g0[1] = ldsOff;                                          // lds_addr
  g0[2] = (unsigned)ga;                                    // global_addr lo
  g0[3] = (unsigned)((ga >> 32) & 0x01FFFFFFu) | (2u << 30);  // ga hi | type=2
  v8u g1;
  g1[0] = (1u << 16) | (1u << 20) | (padIntCode << 22) | (padAmtCode << 25);
  g1[1] = (TD0 & 0xFFFFu) << 16;                           // tensor_dim0 lo16
  g1[2] = (TD0 >> 16) | ((TD1 & 0xFFFFu) << 16);           // td0 hi | td1 lo
  g1[3] = (TD1 >> 16) | (tileW << 16);                     // td1 hi | tile_dim0
  g1[4] = tileH;                                           // tile_dim1 (dim2=0)
  g1[5] = strideElems;                                     // dim0 stride lo32
  g1[6] = 0u;
  g1[7] = 0u;
  asm volatile("tensor_load_to_lds %0, %1" : : "s"(g0), "s"(g1) : "memory");
}

// ---------------------------------------------------------------------------
// f32 -> bf16 conversion (vectorized by 4)
// ---------------------------------------------------------------------------
__global__ __launch_bounds__(256) void cvt_kernel(const float* __restrict__ in,
                                                  unsigned short* __restrict__ out,
                                                  int n4) {
  int i = blockIdx.x * 256 + threadIdx.x;
  if (i >= n4) return;
  float4 f = ((const float4*)in)[i];
  uint2 p;
  p.x = (unsigned)f2bf(f.x) | ((unsigned)f2bf(f.y) << 16);
  p.y = (unsigned)f2bf(f.z) | ((unsigned)f2bf(f.w) << 16);
  ((uint2*)out)[i] = p;
}

// ---------------------------------------------------------------------------
// Tiled WMMA GEMM: C = A(bf16, MxK) * B(bf16, KxN) + bias, optional ReLU,
// bf16 or f32 output. 128x128 tile per 256-thread block (8 waves 4x2), each
// wave 32x64 via 2x4 accumulators. K-step 32. A/B staged by TDM, double
// buffered so DMA of slab k+1 overlaps WMMA on slab k.
// ---------------------------------------------------------------------------
template <bool RELU, bool OUTBF16>
__global__ __launch_bounds__(256) void gemm_bf16_kernel(
    const unsigned short* __restrict__ A, const unsigned short* __restrict__ B,
    const float* __restrict__ bias, void* __restrict__ Cout, int M, int N, int K) {
  __shared__ __align__(16) unsigned short Al[2][128 * 40];  // 128r x 32K (+16B pad)
  __shared__ __align__(16) unsigned short Bl[2][32 * 136];  // 32K x 128N (+16B pad)

  const int tid  = threadIdx.x;
  const int lane = tid & 31;
  const int wave = tid >> 5;
  const int wm   = wave & 3;   // 4 waves along M
  const int wn   = wave >> 2;  // 2 waves along N

  const int tilesN = N >> 7;
  const int m0 = (blockIdx.x / tilesN) << 7;
  const int n0 = (blockIdx.x % tilesN) << 7;

  v8f acc[2][4];
#pragma unroll
  for (int mf = 0; mf < 2; ++mf)
#pragma unroll
    for (int nf = 0; nf < 4; ++nf) acc[mf][nf] = v8f_zero();

  if (wave == 0) {  // prologue DMA: slab 0 -> buffer 0
    tdm_load_2d(A + (size_t)m0 * K, (unsigned)(size_t)&Al[0][0], 32, 128, K, 3, 3);
    tdm_load_2d(B + n0, (unsigned)(size_t)&Bl[0][0], 128, 32, N, 5, 3);
  }

  for (int k0 = 0; k0 < K; k0 += 32) {
    const int cur = (k0 >> 5) & 1;
    if (wave == 0) {
      if (k0 + 32 < K) {
        const int nxt = cur ^ 1;
        tdm_load_2d(A + (size_t)m0 * K + (k0 + 32),
                    (unsigned)(size_t)&Al[nxt][0], 32, 128, K, 3, 3);
        tdm_load_2d(B + (size_t)(k0 + 32) * N + n0,
                    (unsigned)(size_t)&Bl[nxt][0], 128, 32, N, 5, 3);
        __builtin_amdgcn_s_wait_tensorcnt(2);  // slab `cur` landed
      } else {
        __builtin_amdgcn_s_wait_tensorcnt(0);
      }
    }
    __syncthreads();  // publish slab `cur` to all waves

    v16bf af[2], bfm[4];
#pragma unroll
    for (int mf = 0; mf < 2; ++mf)
      af[mf] = fragA16(&Al[cur][0] + (wm * 32 + mf * 16) * 40, 40, 0, lane);
#pragma unroll
    for (int nf = 0; nf < 4; ++nf)
      bfm[nf] = fragB16(&Bl[cur][0], 136, 0, wn * 64 + nf * 16, lane);
#pragma unroll
    for (int nf = 0; nf < 4; ++nf) {
      acc[0][nf] = wmma_bf16(af[0], bfm[nf], acc[0][nf]);
      acc[1][nf] = wmma_bf16(af[1], bfm[nf], acc[1][nf]);
    }
    __syncthreads();  // all reads of slab `cur` done before it is overwritten
  }

  // Epilogue: C layout reg r, lane l -> row = r + 8*(l>>4), col = l&15.
#pragma unroll
  for (int mf = 0; mf < 2; ++mf)
#pragma unroll
    for (int nf = 0; nf < 4; ++nf) {
      const int col = n0 + wn * 64 + nf * 16 + (lane & 15);
      const float bv = bias[col];
#pragma unroll
      for (int r = 0; r < 8; ++r) {
        const int row = m0 + wm * 32 + mf * 16 + r + 8 * (lane >> 4);
        float v = acc[mf][nf][r] + bv;
        if (RELU) v = fmaxf(v, 0.f);
        if (OUTBF16)
          ((unsigned short*)Cout)[(size_t)row * N + col] = f2bf(v);
        else
          ((float*)Cout)[(size_t)row * N + col] = v;
      }
    }
}

// ---------------------------------------------------------------------------
// Attention (transposed orientation): S^T = K Q^T so the per-block operand (K)
// uses cheap b128 A-gathers and each query lives in a fixed lane.
// scores = (Q K^T) * tril / 8; softmax over FULL row: masked logits are 0 (not
// -inf) and contribute exp(0)=1 to the denominator, so ALL k-blocks are
// visited. Logits are O(1) (0.02-scaled weights), so exp() without running-max
// is exact softmax (shift-invariance) -> no per-block rescale, no in-loop
// reductions. O = P V accumulated per query-row; normalized by 1/l at the end.
// 256 threads = 8 waves, wave owns 16 q-rows (q-block = 128 rows). K/V 64x64
// tiles DMA'd by TDM, double buffered.
// ---------------------------------------------------------------------------
__global__ __launch_bounds__(256) void attn_kernel(
    const unsigned short* __restrict__ qkv, unsigned short* __restrict__ ctx) {
  const int T = 2048, LD = 3072, E = 1024;
  __shared__ __align__(16) unsigned short Kl[2][64 * 72];  // keys x d (+16B pad)
  __shared__ __align__(16) unsigned short Vl[2][64 * 72];
  __shared__ __align__(16) unsigned short Pl[8][64 * 24];  // per-wave P^T tile

  const int tid  = threadIdx.x;
  const int lane = tid & 31;
  const int wq   = tid >> 5;   // 8 waves
  const int half = lane >> 4;

  const int qb = blockIdx.x & 15;  // T/128 = 16 q-blocks
  const int bh = blockIdx.x >> 4;
  const int b  = bh >> 4;          // H = 16
  const int h  = bh & 15;

  const int q0 = qb * 128 + wq * 16;
  const size_t rowBase = (size_t)b * T;
  const int qcol = h * 192;
  const int kcol = qcol + 64;
  const int vcol = qcol + 128;

  // Q^T B-operand fragments, gathered once from global.
  v16bf qtf[2];
#pragma unroll
  for (int kc = 0; kc < 2; ++kc)
    qtf[kc] = fragBT16g(qkv + (rowBase + q0) * LD + qcol, LD, kc * 32, lane);

  float lsum = 0.f;   // per-lane partial softmax denominator (query = lane&15)
  v8f o[4];
#pragma unroll
  for (int nf = 0; nf < 4; ++nf) o[nf] = v8f_zero();

  if (wq == 0) {  // prologue DMA: key-block 0 -> buffer 0
    tdm_load_2d(qkv + rowBase * LD + kcol, (unsigned)(size_t)&Kl[0][0], 64, 64, LD, 4, 3);
    tdm_load_2d(qkv + rowBase * LD + vcol, (unsigned)(size_t)&Vl[0][0], 64, 64, LD, 4, 3);
  }

  for (int kb = 0; kb < 32; ++kb) {  // ALL key blocks (multiplicative mask!)
    const int k0 = kb * 64;
    const int cur = kb & 1;
    if (wq == 0) {
      if (kb + 1 < 32) {
        const int nxt = cur ^ 1;
        tdm_load_2d(qkv + (rowBase + k0 + 64) * LD + kcol,
                    (unsigned)(size_t)&Kl[nxt][0], 64, 64, LD, 4, 3);
        tdm_load_2d(qkv + (rowBase + k0 + 64) * LD + vcol,
                    (unsigned)(size_t)&Vl[nxt][0], 64, 64, LD, 4, 3);
        __builtin_amdgcn_s_wait_tensorcnt(2);  // block `cur` landed
      } else {
        __builtin_amdgcn_s_wait_tensorcnt(0);
      }
    }
    __syncthreads();

    const unsigned short* Kb = &Kl[cur][0];
    const unsigned short* Vb = &Vl[cur][0];

    // S^T = K Q^T : keys along M (4 tiles), queries along N (lanes)
    v16bf kfA[4][2];
#pragma unroll
    for (int mt = 0; mt < 4; ++mt)
#pragma unroll
      for (int kc = 0; kc < 2; ++kc)
        kfA[mt][kc] = fragA16(Kb + mt * 16 * 72, 72, kc * 32, lane);

    v8f s[4];
#pragma unroll
    for (int mt = 0; mt < 4; ++mt) {
      s[mt] = v8f_zero();
      s[mt] = wmma_bf16(kfA[mt][0], qtf[0], s[mt]);
      s[mt] = wmma_bf16(kfA[mt][1], qtf[1], s[mt]);
    }

    // mask*scale -> p = exp(.); masked -> exp(0)=1 stays in denominator
    const int qg = q0 + (lane & 15);
#pragma unroll
    for (int mt = 0; mt < 4; ++mt)
#pragma unroll
      for (int r = 0; r < 8; ++r) {
        const int kg = k0 + mt * 16 + r + 8 * half;
        const float z = (kg <= qg) ? s[mt][r] * 0.125f : 0.f;
        const float p = __expf(z);
        s[mt][r] = p;
        lsum += p;
      }

    // P^T (keys x queries, C layout) -> per-wave LDS
    unsigned short* pw = &Pl[wq][0];
#pragma unroll
    for (int mt = 0; mt < 4; ++mt)
#pragma unroll
      for (int r = 0; r < 8; ++r)
        pw[(mt * 16 + r + 8 * half) * 24 + (lane & 15)] = f2bf(s[mt][r]);
    __syncthreads();

    // O += P V : P A-frags strided from P^T (2/block), V B-frags strided u16 gather
    v16bf pfr[2];
#pragma unroll
    for (int kc = 0; kc < 2; ++kc) pfr[kc] = fragAT16(pw, 24, kc * 32, lane);
    v16bf vfB[4][2];
#pragma unroll
    for (int nf = 0; nf < 4; ++nf)
#pragma unroll
      for (int kc = 0; kc < 2; ++kc)
        vfB[nf][kc] = fragB16(Vb, 72, kc * 32, nf * 16, lane);
#pragma unroll
    for (int nf = 0; nf < 4; ++nf) {
      o[nf] = wmma_bf16(pfr[0], vfB[nf][0], o[nf]);
      o[nf] = wmma_bf16(pfr[1], vfB[nf][1], o[nf]);
    }
    __syncthreads();  // reads of buffer `cur` done before next DMA overwrites
  }

  // finalize: combine half-lane partial sums (one shuffle), normalize, store.
  const float ltot = lsum + __shfl_xor(lsum, 16, 32);
  const float linv = 1.f / ltot;  // valid for query = lane&15
#pragma unroll
  for (int r = 0; r < 8; ++r) {
    const float lr = __shfl(linv, r + 8 * half, 32);  // l for query r+8*half
    const size_t row = rowBase + q0 + r + 8 * half;
#pragma unroll
    for (int nf = 0; nf < 4; ++nf)
      ctx[row * E + h * 64 + nf * 16 + (lane & 15)] = f2bf(o[nf][r] * lr);
  }
}

// ---------------------------------------------------------------------------
// Launch pipeline
// ---------------------------------------------------------------------------
extern "C" void kernel_launch(void* const* d_in, const int* in_sizes, int n_in,
                              void* d_out, int out_size, void* d_ws, size_t ws_size,
                              hipStream_t stream) {
  (void)in_sizes; (void)n_in; (void)out_size; (void)ws_size;
  const float* x    = (const float*)d_in[0];
  const float* Wqkv = (const float*)d_in[1];
  const float* bqkv = (const float*)d_in[2];
  const float* W1   = (const float*)d_in[3];
  const float* b1   = (const float*)d_in[4];
  const float* W2   = (const float*)d_in[5];
  const float* b2   = (const float*)d_in[6];

  const int B = 2, T = 2048, E = 1024;
  const int M = B * T, N3 = 3 * E;

  char* ws = (char*)d_ws;
  size_t off = 0;
  auto alloc = [&](size_t bytes) -> void* {
    void* p = ws + off;
    off += (bytes + 255) & ~(size_t)255;
    return p;
  };
  unsigned short* xb    = (unsigned short*)alloc((size_t)M * E * 2);
  unsigned short* wqkvb = (unsigned short*)alloc((size_t)E * N3 * 2);
  unsigned short* w1b   = (unsigned short*)alloc((size_t)E * E * 2);
  unsigned short* w2b   = (unsigned short*)alloc((size_t)E * E * 2);
  unsigned short* qkvb  = (unsigned short*)alloc((size_t)M * N3 * 2);
  unsigned short* ctxb  = (unsigned short*)alloc((size_t)M * E * 2);
  unsigned short* h1b   = (unsigned short*)alloc((size_t)M * E * 2);

  auto cvt = [&](const float* in, unsigned short* out, int n) {
    int n4 = n / 4;
    cvt_kernel<<<dim3((n4 + 255) / 256), dim3(256), 0, stream>>>(in, out, n4);
  };
  cvt(x, xb, M * E);
  cvt(Wqkv, wqkvb, E * N3);
  cvt(W1, w1b, E * E);
  cvt(W2, w2b, E * E);

  // qkv = x @ Wqkv + bqkv  (bf16 out)
  gemm_bf16_kernel<false, true>
      <<<dim3((M / 128) * (N3 / 128)), dim3(256), 0, stream>>>(
          xb, wqkvb, bqkv, qkvb, M, N3, E);

  // attention -> ctx (bf16, (B,T,E))
  attn_kernel<<<dim3(B * 16 * (T / 128)), dim3(256), 0, stream>>>(qkvb, ctxb);

  // h1 = relu(ctx @ W1 + b1)  (bf16 out)
  gemm_bf16_kernel<true, true>
      <<<dim3((M / 128) * (E / 128)), dim3(256), 0, stream>>>(
          ctxb, w1b, b1, h1b, M, E, E);

  // out = h1 @ W2 + b2  (f32 out)
  gemm_bf16_kernel<false, false>
      <<<dim3((M / 128) * (E / 128)), dim3(256), 0, stream>>>(
          h1b, w2b, b2, d_out, M, E, E);
}
